// DriverModel_80461917323828
// MI455X (gfx1250) — matrix operand order repaired and checked by reading, not compiled, not run on hardware
//
#include <hip/hip_runtime.h>

// ---------------------------------------------------------------------------
// DLRM forward on MI455X (gfx1250, wave32).
// All matrix math via v_wmma_f32_16x16x32_bf16 (bf16 operands, fp32 accum),
// everything padded in workspace so WMMA inner loops have zero bounds checks.
// GEMM v3: block-cooperative 128Mx64N macro-tile. The 8 waves of a workgroup
// stage the Kx64 B panel into LDS once via global_load_async_to_lds_b128
// (ASYNCcnt-tracked), then each wave computes 2 M-tiles x 4 N-tiles:
// per k-step 4 global A loads + 64 LDS B element loads + 8 WMMAs.
// ---------------------------------------------------------------------------

typedef __attribute__((ext_vector_type(8)))  float  v8f;
typedef __attribute__((ext_vector_type(8)))  __bf16 v8bf;
typedef __attribute__((ext_vector_type(16))) __bf16 v16bf;
typedef __attribute__((ext_vector_type(4)))  int    i4v;

#define BATCH      16384
#define NUM_DENSE  13
#define NUM_SPARSE 26
#define VOCAB      100000
#define EMB        64
#define NPAIRS     325             // 26*25/2
#define FEAT_RAW   (64 + NPAIRS)   // 389
#define FEAT_PAD   416             // 13*32, K-padded for top GEMM
#define DENSE_PAD  32
#define SP_PAD     32              // 26 -> 32 rows
#define TOP_OUT    256

// ------------------- gfx1250 async global->LDS detection --------------------
#if defined(__has_builtin)
#if __has_builtin(__builtin_amdgcn_global_load_async_to_lds_b128)
#define HAVE_ASYNC_LDS 1
#endif
#endif
#ifndef HAVE_ASYNC_LDS
#define HAVE_ASYNC_LDS 0
#endif

#if HAVE_ASYNC_LDS
// signature (from clang diagnostic): (i4v AS1*, i4v AS3*, imm int, imm int)
__device__ __forceinline__ void async_copy16(const __bf16* gsrc, __bf16* ldst) {
  __builtin_amdgcn_global_load_async_to_lds_b128(
      (__attribute__((address_space(1))) i4v*)(i4v*)const_cast<__bf16*>(gsrc),
      (__attribute__((address_space(3))) i4v*)(i4v*)ldst, 0, 0);
}
#if __has_builtin(__builtin_amdgcn_s_wait_asynccnt)
#define ASYNC_WAIT() __builtin_amdgcn_s_wait_asynccnt(0)
#else
#define ASYNC_WAIT() asm volatile("s_wait_asynccnt 0x0" ::: "memory")
#endif
#endif

// --------------------- fp32 -> bf16 convert with zero pad -------------------
__global__ void cvt_pad_kernel(const float* __restrict__ src, int srcR, int srcC,
                               __bf16* __restrict__ dst, int dstR, int dstC) {
  int i = blockIdx.x * blockDim.x + threadIdx.x;
  if (i >= dstR * dstC) return;
  int r = i / dstC, c = i % dstC;
  float v = (r < srcR && c < srcC) ? src[r * srcC + c] : 0.0f;
  dst[i] = (__bf16)v;
}

// --------------------- strided zero fill (padding regions) ------------------
__global__ void zero_pad_kernel(__bf16* __restrict__ base, int outer,
                                int ostride, int offset, int count) {
  int i = blockIdx.x * blockDim.x + threadIdx.x;
  if (i >= outer * count) return;
  int o = i / count, j = i % count;
  base[(size_t)o * ostride + offset + j] = (__bf16)0.0f;
}

// ------------------------------ gather --------------------------------------
// one wave per (sample, table): 32 lanes x float2 = 64 floats.
// Writes row-major padded emb [B][32][64] and transposed embT [B][64][32].
__global__ void gather_emb_kernel(const float* __restrict__ tables,
                                  const int* __restrict__ idx,
                                  __bf16* __restrict__ emb,
                                  __bf16* __restrict__ embT, int rows) {
  int row  = blockIdx.x * (blockDim.x >> 5) + (threadIdx.x >> 5);
  int lane = threadIdx.x & 31;
  if (row >= rows) return;
  int b = row / NUM_SPARSE;
  int t = row % NUM_SPARSE;
  int id = idx[row];
  const float2* src =
      (const float2*)(tables + ((size_t)t * VOCAB + (size_t)id) * EMB);
  float2 v = src[lane];
  __bf16 lo = (__bf16)v.x, hi = (__bf16)v.y;
  __bf16* drow = emb + ((size_t)b * SP_PAD + t) * EMB;
  drow[2 * lane]     = lo;
  drow[2 * lane + 1] = hi;
  __bf16* tb = embT + (size_t)b * EMB * SP_PAD + t;
  tb[(2 * lane) * SP_PAD]     = lo;
  tb[(2 * lane + 1) * SP_PAD] = hi;
}

// ------------------------------ WMMA helpers --------------------------------
__device__ __forceinline__ v16bf load_a_frag(const __bf16* arow, int k0, int half) {
  v8bf lo = *(const v8bf*)(arow + k0 + 8 * half);
  v8bf hi = *(const v8bf*)(arow + k0 + 16 + 8 * half);
  return __builtin_shufflevector(lo, hi, 0, 1, 2, 3, 4, 5, 6, 7,
                                 8, 9, 10, 11, 12, 13, 14, 15);
}

__device__ __forceinline__ v16bf combine(v8bf lo, v8bf hi) {
  return __builtin_shufflevector(lo, hi, 0, 1, 2, 3, 4, 5, 6, 7,
                                 8, 9, 10, 11, 12, 13, 14, 15);
}

// ------------------------------ generic GEMM --------------------------------
// Block: 256 threads (8 waves) computing a 128(M) x 64(N) macro tile.
// B panel (K x 64) staged once into LDS (row stride 72 elems = 144B to spread
// banks). Each wave: 2 M-tiles x 4 N-tiles. M%128==0, N%64==0, K%32==0.
#define LDS_STRIDE 72
__global__ void gemm_bf16_wmma_kernel(const __bf16* __restrict__ A, int lda,
                                      const __bf16* __restrict__ Bm, int ldb,
                                      const float* __restrict__ bias,
                                      int M, int N, int K,
                                      __bf16* __restrict__ obf, int ldob, int ocol,
                                      float* __restrict__ ofl, int ldof,
                                      int relu, int nGroups /* N/64 */) {
  extern __shared__ char smem_raw[];
  __bf16* ldsB = (__bf16*)smem_raw;

  int tid  = threadIdx.x;
  int lane = tid & 31;
  int wv   = tid >> 5;
  int mb   = blockIdx.x / nGroups;
  int ng   = blockIdx.x % nGroups;

  // ---- stage B panel (K x 64 bf16) into LDS: K*8 16-byte chunks ----
  for (int i = tid; i < K * 8; i += 256) {
    int k = i >> 3, chunk = i & 7;
    const __bf16* gsrc = Bm + (size_t)k * ldb + ng * 64 + chunk * 8;
    __bf16* ldst = ldsB + k * LDS_STRIDE + chunk * 8;
#if HAVE_ASYNC_LDS
    async_copy16(gsrc, ldst);
#else
    *(uint4*)ldst = *(const uint4*)gsrc;
#endif
  }
#if HAVE_ASYNC_LDS
  ASYNC_WAIT();
#endif
  __syncthreads();

  // ---- compute: wave wv owns M rows [mb*128 + wv*32, +32) ----
  int half = lane >> 4;
  const __bf16* arow0 = A + (size_t)(mb * 128 + wv * 32 + (lane & 15)) * lda;
  const __bf16* arow1 = arow0 + (size_t)16 * lda;

  v8f acc[2][4] = {};
  for (int k0 = 0; k0 < K; k0 += 32) {
    v16bf a0 = load_a_frag(arow0, k0, half);
    v16bf a1 = load_a_frag(arow1, k0, half);
    const __bf16* bbase = ldsB + k0 * LDS_STRIDE + (lane & 15);
#pragma unroll
    for (int t = 0; t < 4; ++t) {
      v16bf b;
#pragma unroll
      for (int s = 0; s < 16; ++s)
        b[s] = bbase[(8 * half + (s & 7) + 16 * (s >> 3)) * LDS_STRIDE + t * 16];
      acc[0][t] = __builtin_amdgcn_wmma_f32_16x16x32_bf16(
          false, a0, false, b, (short)0, acc[0][t], false, false);
      acc[1][t] = __builtin_amdgcn_wmma_f32_16x16x32_bf16(
          false, a1, false, b, (short)0, acc[1][t], false, false);
    }
  }

#pragma unroll
  for (int mt = 0; mt < 2; ++mt) {
#pragma unroll
    for (int t = 0; t < 4; ++t) {
      int col = ng * 64 + t * 16 + (lane & 15);
      float bv = bias[col];
#pragma unroll
      for (int i = 0; i < 8; ++i) {
        int row = mb * 128 + wv * 32 + mt * 16 + i + 8 * half;
        float v = acc[mt][t][i] + bv;
        if (relu) v = fmaxf(v, 0.0f);
        if (ofl) ofl[(size_t)row * ldof + col] = v;
        else     obf[(size_t)row * ldob + ocol + col] = (__bf16)v;
      }
    }
  }
}

// ------------------------------ interaction ---------------------------------
// gram = E * E^T, E = 26x64 padded to 32x64 (emb) with E^T in embT [64][32].
// One wave per (sample, tile-pair); pairs (0,0),(0,1),(1,1) cover the triu.
__global__ void interact_wmma_kernel(const __bf16* __restrict__ emb,
                                     const __bf16* __restrict__ embT,
                                     __bf16* __restrict__ feat, int Bsz) {
  int lane = threadIdx.x & 31;
  int wave = (int)((blockIdx.x * blockDim.x + threadIdx.x) >> 5);
  if (wave >= Bsz * 3) return;
  int b = wave / 3, t = wave % 3;
  int ti = (t == 2) ? 1 : 0;
  int tj = (t == 0) ? 0 : 1;
  const __bf16* E  = emb  + (size_t)b * SP_PAD * EMB;
  const __bf16* Et = embT + (size_t)b * EMB * SP_PAD;
  int half = lane >> 4;
  const __bf16* arow = E + (size_t)(ti * 16 + (lane & 15)) * EMB;

  v8f acc = {};
#pragma unroll
  for (int k0 = 0; k0 < EMB; k0 += 32) {
    v16bf a = load_a_frag(arow, k0, half);
    v16bf bb = load_a_frag(E + (size_t)(tj * 16 + (lane & 15)) * EMB, k0, half);
    acc = __builtin_amdgcn_wmma_f32_16x16x32_bf16(false, a, false, bb,
                                                  (short)0, acc, false, false);
  }

#pragma unroll
  for (int i = 0; i < 8; ++i) {
    int gi = ti * 16 + i + 8 * half;
    int gj = tj * 16 + (lane & 15);
    if (gi < NUM_SPARSE && gj < NUM_SPARSE && gi < gj) {
      int pos = gi * (NUM_SPARSE - 1) - (gi * (gi - 1)) / 2 + (gj - gi - 1);
      feat[(size_t)b * FEAT_PAD + 64 + pos] = (__bf16)acc[i];
    }
  }
}

// ------------------------------ launch --------------------------------------
static inline int cdiv(int a, int b) { return (a + b - 1) / b; }

static void launch_gemm(const __bf16* A, int lda, const __bf16* B, int ldb,
                        const float* bias, int M, int N, int K,
                        __bf16* obf, int ldob, int ocol,
                        float* ofl, int ldof, int relu, hipStream_t stream) {
  int nGroups = N / 64;
  int blocks  = (M / 128) * nGroups;
  size_t lds  = (size_t)K * LDS_STRIDE * sizeof(__bf16);
  gemm_bf16_wmma_kernel<<<blocks, 256, lds, stream>>>(
      A, lda, B, ldb, bias, M, N, K, obf, ldob, ocol, ofl, ldof, relu, nGroups);
}

extern "C" void kernel_launch(void* const* d_in, const int* in_sizes, int n_in,
                              void* d_out, int out_size, void* d_ws, size_t ws_size,
                              hipStream_t stream) {
  const float* dense  = (const float*)d_in[0];
  const int*   sparse = (const int*)d_in[1];
  const float* tables = (const float*)d_in[2];
  const float* W0 = (const float*)d_in[3];
  const float* b0 = (const float*)d_in[4];
  const float* W1 = (const float*)d_in[5];
  const float* b1 = (const float*)d_in[6];
  const float* W2 = (const float*)d_in[7];
  const float* b2 = (const float*)d_in[8];
  const float* Wt = (const float*)d_in[9];
  const float* bt = (const float*)d_in[10];
  float* out = (float*)d_out;

  // workspace carve-up (256B aligned)
  char* p = (char*)d_ws;
  auto carve = [&](size_t bytes) {
    char* r = p;
    p += (bytes + 255) & ~(size_t)255;
    return r;
  };
  __bf16* emb    = (__bf16*)carve((size_t)BATCH * SP_PAD * EMB * 2);
  __bf16* embT   = (__bf16*)carve((size_t)BATCH * EMB * SP_PAD * 2);
  __bf16* denseb = (__bf16*)carve((size_t)BATCH * DENSE_PAD * 2);
  __bf16* x0     = (__bf16*)carve((size_t)BATCH * 512 * 2);
  __bf16* x1     = (__bf16*)carve((size_t)BATCH * 256 * 2);
  __bf16* feat   = (__bf16*)carve((size_t)BATCH * FEAT_PAD * 2);
  __bf16* W0b    = (__bf16*)carve((size_t)DENSE_PAD * 512 * 2);
  __bf16* W1b    = (__bf16*)carve((size_t)512 * 256 * 2);
  __bf16* W2b    = (__bf16*)carve((size_t)256 * 64 * 2);
  __bf16* Wtb    = (__bf16*)carve((size_t)FEAT_PAD * TOP_OUT * 2);

  // fp32 -> bf16 (zero padded)
  auto cvt = [&](const float* s, int sr, int sc, __bf16* d, int dr, int dc) {
    cvt_pad_kernel<<<cdiv(dr * dc, 256), 256, 0, stream>>>(s, sr, sc, d, dr, dc);
  };
  cvt(dense, BATCH, NUM_DENSE, denseb, BATCH, DENSE_PAD);
  cvt(W0, NUM_DENSE, 512, W0b, DENSE_PAD, 512);
  cvt(W1, 512, 256, W1b, 512, 256);
  cvt(W2, 256, 64, W2b, 256, 64);
  cvt(Wt, FEAT_RAW, TOP_OUT, Wtb, FEAT_PAD, TOP_OUT);

  // zero the padding regions written by neither gather nor GEMMs
  zero_pad_kernel<<<cdiv(BATCH * 6 * EMB, 256), 256, 0, stream>>>(
      emb, BATCH, SP_PAD * EMB, NUM_SPARSE * EMB, 6 * EMB);       // rows 26..31
  zero_pad_kernel<<<cdiv(BATCH * EMB * 6, 256), 256, 0, stream>>>(
      embT, BATCH * EMB, SP_PAD, NUM_SPARSE, 6);                  // cols 26..31
  zero_pad_kernel<<<cdiv(BATCH * (FEAT_PAD - FEAT_RAW), 256), 256, 0, stream>>>(
      feat, BATCH, FEAT_PAD, FEAT_RAW, FEAT_PAD - FEAT_RAW);      // cols 389..415

  // embedding gather (~110MB random reads; memory-bound)
  {
    int rows = BATCH * NUM_SPARSE;
    gather_emb_kernel<<<cdiv(rows, 8), 256, 0, stream>>>(tables, sparse,
                                                         emb, embT, rows);
  }

  // bottom MLP (ReLU); layer 2 writes into feature buffer cols [0,64)
  launch_gemm(denseb, DENSE_PAD, W0b, 512, b0, BATCH, 512, DENSE_PAD,
              x0, 512, 0, nullptr, 0, /*relu=*/1, stream);
  launch_gemm(x0, 512, W1b, 256, b1, BATCH, 256, 512,
              x1, 256, 0, nullptr, 0, /*relu=*/1, stream);
  launch_gemm(x1, 256, W2b, 64, b2, BATCH, 64, 256,
              feat, FEAT_PAD, 0, nullptr, 0, /*relu=*/1, stream);

  // pairwise interaction -> feature buffer cols [64,389)
  interact_wmma_kernel<<<cdiv(BATCH * 3, 8), 256, 0, stream>>>(emb, embT,
                                                               feat, BATCH);

  // top GEMM -> fp32 output (K padded to 416 with zeros)
  launch_gemm(feat, FEAT_PAD, Wtb, TOP_OUT, bt, BATCH, TOP_OUT, FEAT_PAD,
              nullptr, 0, 0, out, TOP_OUT, /*relu=*/0, stream);
}
